// Net_71889162600901
// MI455X (gfx1250) — compile-verified, matrix operand order checked by reading, not run on hardware
//
#include <hip/hip_runtime.h>
#include <hip/hip_bf16.h>

typedef _Float16 h8  __attribute__((ext_vector_type(8)));
typedef _Float16 h16 __attribute__((ext_vector_type(16)));
typedef float    f8  __attribute__((ext_vector_type(8)));

#define NHEAD 8
#define HIDC  64
#define HDIM  512   /* NHEAD*HIDC */
#define FINC  512
#define OUTD  128
#define SLOPE 0.2f

// ---------------- elementwise converts ----------------
__global__ void k_f32_to_f16(const float* __restrict__ x, _Float16* __restrict__ y, int n) {
  int i = blockIdx.x * blockDim.x + threadIdx.x;
  if (i < n) y[i] = (_Float16)x[i];
}

// W [K,N] f32 row-major  ->  Wt [N,K] f16 row-major
__global__ void k_transpose_to_f16(const float* __restrict__ W, _Float16* __restrict__ Wt,
                                   int K, int N) {
  int i = blockIdx.x * blockDim.x + threadIdx.x;
  if (i >= K * N) return;
  int k = i / N, n = i - k * N;
  Wt[(size_t)n * K + k] = (_Float16)W[i];
}

// out(f32) + bias -> relu -> f16
__global__ void k_bias_relu_h(const float* __restrict__ x, const float* __restrict__ bias,
                              _Float16* __restrict__ y, int n, int C) {
  int i = blockIdx.x * blockDim.x + threadIdx.x;
  if (i >= n) return;
  int c = i % C;
  float v = x[i] + bias[c];
  y[i] = (_Float16)(v > 0.f ? v : 0.f);
}

// ---------------- WMMA GEMM ----------------
// C[M,N] = A[M,K] (f16,row-major) x Bt[N,K]^T (f16,row-major) (+bias)(+relu), f32 out.
// One wave computes a 16x64 tile: 4 accumulators share the A fragment.
// A-fragment layout (16-bit A 16x32): lanes 0-15 row M=lane, K = {0..7,16..23};
// lanes 16-31 row M=lane-16, K = {8..15,24..31}. B fragment mirrors with N per lane.
// Out-of-range M rows are clamped to row 0 on load (their garbage products are
// never stored), so no fragment zeroing is needed in the hot loop.
__global__ void k_wmma_gemm(const _Float16* __restrict__ A,
                            const _Float16* __restrict__ Bt,
                            const float* __restrict__ bias,
                            float* __restrict__ C,
                            int M, int N, int K, int relu) {
  const int lane = threadIdx.x & 31;
  const int lm = lane & 15;
  const int hiHalf = lane >> 4;
  const int kb = hiHalf * 8;
  const int m0 = blockIdx.y * 16;
  const int n0 = blockIdx.x * 64;

  f8 acc[4] = {};
  const int mA = m0 + lm;
  const size_t arow = (size_t)(mA < M ? mA : 0) * K;

  union frag { h8 v2[2]; h16 v; };

  for (int kk = 0; kk < K; kk += 32) {
    frag a;
    a.v2[0] = *(const h8*)(A + arow + kk + kb);
    a.v2[1] = *(const h8*)(A + arow + kk + kb + 16);
#pragma unroll
    for (int t = 0; t < 4; ++t) {
      size_t brow = (size_t)(n0 + t * 16 + lm) * K;
      frag b;
      b.v2[0] = *(const h8*)(Bt + brow + kk + kb);
      b.v2[1] = *(const h8*)(Bt + brow + kk + kb + 16);
      acc[t] = __builtin_amdgcn_wmma_f32_16x16x32_f16(
          /*neg_a=*/false, a.v, /*neg_b=*/false, b.v,
          /*c_mod=*/(short)0, acc[t], /*reuse_a=*/false, /*reuse_b=*/false);
    }
  }

  // C/D layout: lanes 0-15 -> M=m0+r, lanes 16-31 -> M=m0+r+8; N = n0+t*16+lm.
#pragma unroll
  for (int t = 0; t < 4; ++t) {
    int n = n0 + t * 16 + lm;
    if (n >= N) continue;
#pragma unroll
    for (int r = 0; r < 8; ++r) {
      int m = m0 + r + hiHalf * 8;
      if (m < M) {
        float v = acc[t][r];
        if (bias) v += bias[n];
        if (relu) v = v > 0.f ? v : 0.f;
        C[(size_t)m * N + n] = v;
      }
    }
  }
}

// ---------------- attention pieces ----------------
// per (node, head): dot of h[n,h,:] with a_src/a_dst rows
__global__ void k_attn_scores(const float* __restrict__ Hf,
                              const float* __restrict__ asrc, const float* __restrict__ adst,
                              float* __restrict__ s_out, float* __restrict__ d_out_, int Nn) {
  int t = blockIdx.x * blockDim.x + threadIdx.x;
  if (t >= Nn * NHEAD) return;
  int n = t >> 3, h = t & 7;
  const float* hp = Hf + (size_t)n * HDIM + h * HIDC;
  float s1 = 0.f, s2 = 0.f;
  for (int c = 0; c < HIDC; ++c) {
    float v = hp[c];
    s1 += v * asrc[h * HIDC + c];
    s2 += v * adst[h * HIDC + c];
  }
  s_out[t] = s1;
  d_out_[t] = s2;
}

__device__ __forceinline__ unsigned enc_f32(float x) {
  unsigned v = __float_as_uint(x);
  return (v & 0x80000000u) ? ~v : (v | 0x80000000u);
}
__device__ __forceinline__ float dec_f32(unsigned u) {
  return (u & 0x80000000u) ? __uint_as_float(u & 0x7FFFFFFFu) : __uint_as_float(~u);
}

// pass 1: segment max of leaky_relu(as[src]+ad[dst]) over dst, via encoded atomicMax
__global__ void k_edge_max(const int* __restrict__ src, const int* __restrict__ dst,
                           int E, int Nn,
                           const float* __restrict__ as_, const float* __restrict__ ad_,
                           unsigned* __restrict__ emax_enc) {
  int t = blockIdx.x * blockDim.x + threadIdx.x;
  int tot = (E + Nn) * NHEAD;
  if (t >= tot) return;
  int e = t >> 3, h = t & 7;
  int s = (e < E) ? src[e] : (e - E);
  int d = (e < E) ? dst[e] : (e - E);
  float x = as_[s * NHEAD + h] + ad_[d * NHEAD + h];
  x = x > 0.f ? x : SLOPE * x;
  atomicMax(&emax_enc[d * NHEAD + h], enc_f32(x));
}

// pass 2: ex = exp(e - max), store + segment-sum denom
__global__ void k_edge_expsum(const int* __restrict__ src, const int* __restrict__ dst,
                              int E, int Nn,
                              const float* __restrict__ as_, const float* __restrict__ ad_,
                              const unsigned* __restrict__ emax_enc,
                              float* __restrict__ ex, float* __restrict__ den) {
  int t = blockIdx.x * blockDim.x + threadIdx.x;
  int tot = (E + Nn) * NHEAD;
  if (t >= tot) return;
  int e = t >> 3, h = t & 7;
  int s = (e < E) ? src[e] : (e - E);
  int d = (e < E) ? dst[e] : (e - E);
  float x = as_[s * NHEAD + h] + ad_[d * NHEAD + h];
  x = x > 0.f ? x : SLOPE * x;
  float v = expf(x - dec_f32(emax_enc[d * NHEAD + h]));
  ex[t] = v;
  atomicAdd(&den[d * NHEAD + h], v);
}

// pass 3: out[dst] += (ex/den)[e,h] * h[src]; one block (512 threads) per edge
__global__ void k_aggregate(const int* __restrict__ src, const int* __restrict__ dst,
                            int E, int Nn,
                            const float* __restrict__ Hf, const float* __restrict__ ex,
                            const float* __restrict__ den, float* __restrict__ out) {
  int e = blockIdx.x;
  int c = threadIdx.x;                       // 0..511
  int s = (e < E) ? src[e] : (e - E);
  int d = (e < E) ? dst[e] : (e - E);
  int h = c >> 6;
  float alpha = ex[(size_t)e * NHEAD + h] / den[(size_t)d * NHEAD + h];
  atomicAdd(&out[(size_t)d * HDIM + c], alpha * Hf[(size_t)s * HDIM + c]);
}

// ---------------- host orchestration ----------------
static inline int cdiv(int a, int b) { return (a + b - 1) / b; }

extern "C" void kernel_launch(void* const* d_in, const int* in_sizes, int n_in,
                              void* d_out, int out_size, void* d_ws, size_t ws_size,
                              hipStream_t stream) {
  (void)in_sizes; (void)n_in; (void)out_size; (void)ws_size;

  const int Mn[3] = {10000, 5000, 10000};
  const int Eg[3] = {160000, 80000, 160000};
  const float* Xin[3] = {(const float*)d_in[0], (const float*)d_in[1], (const float*)d_in[2]};
  const int* EI[3] = {(const int*)d_in[3], (const int*)d_in[4], (const int*)d_in[5]};

  float* outp = (float*)d_out;
  const size_t outOff[3] = {0, 10000ull * OUTD, 10000ull * OUTD + 5000ull * OUTD};

  // workspace layout (sized for the largest graph, reused sequentially)
  char* ws = (char*)d_ws;
  size_t off = 0;
  auto take = [&](size_t bytes) { size_t o = off; off = (off + bytes + 255) & ~(size_t)255; return o; };
  const size_t O_X16 = take(10000ull * FINC * 2);   // f16 node features / gat-out
  const size_t O_WT  = take(512ull * 512 * 2);      // W^T f16
  const size_t O_H   = take(10000ull * HDIM * 4);   // h = XW (f32)
  const size_t O_AS  = take(10000ull * NHEAD * 4);
  const size_t O_AD  = take(10000ull * NHEAD * 4);
  const size_t O_EMX = take(10000ull * NHEAD * 4);  // encoded max
  const size_t O_DEN = take(10000ull * NHEAD * 4);
  const size_t O_EX  = take(170000ull * NHEAD * 4);
  const size_t O_OUT = take(10000ull * HDIM * 4);   // GAT aggregate (f32)
  const size_t O_W1T = take(64ull * 512 * 2);
  const size_t O_W2T = take(64ull * 64 * 2);
  const size_t O_W3T = take(128ull * 64 * 2);
  const size_t O_X1  = take(10000ull * HIDC * 4);
  const size_t O_X1H = take(10000ull * HIDC * 2);
  const size_t O_X2  = take(10000ull * HIDC * 4);
  const size_t O_X2H = take(10000ull * HIDC * 2);

  for (int g = 0; g < 3; ++g) {
    const int M = Mn[g], E = Eg[g], Et = E + M;
    const float* W    = (const float*)d_in[6 + 4 * g];
    const float* asrc = (const float*)d_in[7 + 4 * g];
    const float* adst = (const float*)d_in[8 + 4 * g];
    const float* bg   = (const float*)d_in[9 + 4 * g];
    const float* w1 = (const float*)d_in[18 + 6 * g]; const float* b1 = (const float*)d_in[19 + 6 * g];
    const float* w2 = (const float*)d_in[20 + 6 * g]; const float* b2 = (const float*)d_in[21 + 6 * g];
    const float* w3 = (const float*)d_in[22 + 6 * g]; const float* b3 = (const float*)d_in[23 + 6 * g];
    const int* srcI = EI[g];
    const int* dstI = EI[g] + E;

    _Float16* x16 = (_Float16*)(ws + O_X16);
    _Float16* wt  = (_Float16*)(ws + O_WT);
    float* Hf   = (float*)(ws + O_H);
    float* as_  = (float*)(ws + O_AS);
    float* ad_  = (float*)(ws + O_AD);
    unsigned* emx = (unsigned*)(ws + O_EMX);
    float* den  = (float*)(ws + O_DEN);
    float* ex   = (float*)(ws + O_EX);
    float* gout = (float*)(ws + O_OUT);
    _Float16* w1t = (_Float16*)(ws + O_W1T);
    _Float16* w2t = (_Float16*)(ws + O_W2T);
    _Float16* w3t = (_Float16*)(ws + O_W3T);
    float* x1 = (float*)(ws + O_X1);  _Float16* x1h = (_Float16*)(ws + O_X1H);
    float* x2 = (float*)(ws + O_X2);  _Float16* x2h = (_Float16*)(ws + O_X2H);

    // ---- h = X @ W  (f16 WMMA, f32 accumulate) ----
    {
      int n = M * FINC;
      k_f32_to_f16<<<cdiv(n, 256), 256, 0, stream>>>(Xin[g], x16, n);
      k_transpose_to_f16<<<cdiv(512 * 512, 256), 256, 0, stream>>>(W, wt, FINC, HDIM);
      dim3 grid(HDIM / 64, cdiv(M, 16));
      k_wmma_gemm<<<grid, 32, 0, stream>>>(x16, wt, nullptr, Hf, M, HDIM, FINC, 0);
    }

    // ---- attention scores, segment softmax, aggregate ----
    k_attn_scores<<<cdiv(M * NHEAD, 256), 256, 0, stream>>>(Hf, asrc, adst, as_, ad_, M);
    hipMemsetAsync(emx, 0, (size_t)M * NHEAD * 4, stream);   // enc(x)=0 is -inf sentinel
    hipMemsetAsync(den, 0, (size_t)M * NHEAD * 4, stream);
    hipMemsetAsync(gout, 0, (size_t)M * HDIM * 4, stream);
    k_edge_max<<<cdiv(Et * NHEAD, 256), 256, 0, stream>>>(srcI, dstI, E, M, as_, ad_, emx);
    k_edge_expsum<<<cdiv(Et * NHEAD, 256), 256, 0, stream>>>(srcI, dstI, E, M, as_, ad_, emx, ex, den);
    k_aggregate<<<Et, HDIM, 0, stream>>>(srcI, dstI, E, M, Hf, ex, den, gout);

    // ---- relu(out + bias) -> f16 (reuses x16) ----
    k_bias_relu_h<<<cdiv(M * HDIM, 256), 256, 0, stream>>>(gout, bg, x16, M * HDIM, HDIM);

    // ---- MLP: 512 -> 64 -> 64 -> 128 ----
    k_transpose_to_f16<<<cdiv(512 * 64, 256), 256, 0, stream>>>(w1, w1t, HDIM, HIDC);
    k_transpose_to_f16<<<cdiv(64 * 64, 256), 256, 0, stream>>>(w2, w2t, HIDC, HIDC);
    k_transpose_to_f16<<<cdiv(64 * 128, 256), 256, 0, stream>>>(w3, w3t, HIDC, OUTD);

    {
      dim3 grid(HIDC / 64, cdiv(M, 16));
      k_wmma_gemm<<<grid, 32, 0, stream>>>(x16, w1t, b1, x1, M, HIDC, HDIM, 1);
      k_f32_to_f16<<<cdiv(M * HIDC, 256), 256, 0, stream>>>(x1, x1h, M * HIDC);
      k_wmma_gemm<<<grid, 32, 0, stream>>>(x1h, w2t, b2, x2, M, HIDC, HIDC, 1);
      k_f32_to_f16<<<cdiv(M * HIDC, 256), 256, 0, stream>>>(x2, x2h, M * HIDC);
      dim3 grid3(OUTD / 64, cdiv(M, 16));
      k_wmma_gemm<<<grid3, 32, 0, stream>>>(x2h, w3t, b3, outp + outOff[g], M, OUTD, HIDC, 0);
    }
  }
}
